// IGASlotPoolingV2_43525198577872
// MI455X (gfx1250) — compile-verified, matrix operand order checked by reading, not run on hardware
//
#include <hip/hip_runtime.h>
#include <hip/hip_bf16.h>
#include <stdint.h>

// ---------------- problem constants (from reference setup_inputs) -----------
#define BB     2
#define NTOK   4096
#define CDIM   512
#define KSLOT  512
#define CGDIM  64
#define CF     128          // combined feature width: [sem 64 | 0.3*geo 64]
#define MTOK   (BB * NTOK)  // 8192
#define KUSE   512

#define NEGF   (-1e9f)
#define BIGF   (1e30f)      // covered value for masked tokens -> gain == 0
#define TAUINV (1.0f / 0.7f)
#define LAMG   0.3f
#define JIT    1e-6f
#define EPSF   1e-8f
#define NRMEPS 1e-12f

// ---------------- CDNA5 vector types ---------------------------------------
typedef __attribute__((ext_vector_type(16))) __bf16       v16bf;
typedef __attribute__((ext_vector_type(8)))  __bf16       v8bf;
typedef __attribute__((ext_vector_type(8)))  float        v8f;
typedef __attribute__((ext_vector_type(4)))  unsigned int u32x4;
typedef __attribute__((ext_vector_type(8)))  int          i32x8;
typedef __attribute__((ext_vector_type(4)))  int          i32x4;

// ---------------------------------------------------------------------------
// f32 -> bf16 elementwise conversion (one-time; removes cvt chains from the
// WMMA hot loop and halves its load bytes).
// ---------------------------------------------------------------------------
__global__ __launch_bounds__(256) void cvt_bf16_kernel(
    const float* __restrict__ in, __bf16* __restrict__ outp, int nelem)
{
  const int i = blockIdx.x * 256 + threadIdx.x;
  if (i < nelem) outp[i] = (__bf16)in[i];
}

// ---------------------------------------------------------------------------
// Generic WMMA GEMM on bf16 operands: C[m,n] = sum_k A[m*Kd+k] * B[n*Kd+k].
// One wave computes a 16x64 strip (1 M-tile x 4 N-tiles, A-frag reused) with
// v_wmma_f32_16x16x32_bf16.  Fragments load as two raw b128s per operand.
// Requires: M % 16 == 0, Nd % 64 == 0, Kd % 32 == 0, 16B-aligned pointers.
// ---------------------------------------------------------------------------
__global__ __launch_bounds__(256) void wmma_gemm_bf16(
    const __bf16* __restrict__ A, const __bf16* __restrict__ Bm,
    float* __restrict__ Cm, int M, int Nd, int Kd)
{
  const int lane = threadIdx.x & 31;
  const int wave = (blockIdx.x * blockDim.x + threadIdx.x) >> 5;
  const int tiles_n4 = Nd >> 6;
  const int tiles_m  = M >> 4;
  if (wave >= tiles_m * tiles_n4) return;
  const int tm   = wave / tiles_n4;
  const int tn4  = wave % tiles_n4;
  const int r    = lane & 15;
  const int half = lane >> 4;

  v8f acc[4] = {};

  // 16-bit A-matrix 16x32 layout: lanes 0-15 -> M rows; per lane:
  //   elems 0..7  hold K = k0 + half*8 + e
  //   elems 8..15 hold K = k0 + 16 + half*8 + e
  const __bf16* ap0 = A + (size_t)(tm * 16 + r) * Kd + half * 8;

  for (int k0 = 0; k0 < Kd; k0 += 32) {
    const __bf16* ap = ap0 + k0;
    const v8bf alo = *(const v8bf*)(ap);
    const v8bf ahi = *(const v8bf*)(ap + 16);
    v16bf a;
#pragma unroll
    for (int e = 0; e < 8; ++e) { a[e] = alo[e]; a[8 + e] = ahi[e]; }
#pragma unroll
    for (int t = 0; t < 4; ++t) {
      const int ncol = (tn4 * 4 + t) * 16 + r;
      const __bf16* bp = Bm + (size_t)ncol * Kd + k0 + half * 8;
      const v8bf blo = *(const v8bf*)(bp);
      const v8bf bhi = *(const v8bf*)(bp + 16);
      v16bf bv;
#pragma unroll
      for (int e = 0; e < 8; ++e) { bv[e] = blo[e]; bv[8 + e] = bhi[e]; }
      acc[t] = __builtin_amdgcn_wmma_f32_16x16x32_bf16(
          /*neg_a=*/false, a, /*neg_b=*/false, bv,
          /*c_mod=*/(short)0, acc[t], /*reuse_a=*/false, /*reuse_b=*/false);
    }
  }

  // 32-bit C/D layout: VGPR i holds M = tm*16 + half*8 + i, N = lane column
#pragma unroll
  for (int t = 0; t < 4; ++t) {
    const int ncol = (tn4 * 4 + t) * 16 + r;
#pragma unroll
    for (int i = 0; i < 8; ++i) {
      const int row = tm * 16 + half * 8 + i;
      Cm[(size_t)row * Nd + ncol] = acc[t][i];
    }
  }
}

// ---------------------------------------------------------------------------
// Token features -> bf16: normalize sem row, geo invariants -> project ->
// normalize -> scale by LAM.  4 tokens per 256-thread block (64 lanes/token).
// ---------------------------------------------------------------------------
__global__ __launch_bounds__(256) void token_feat_kernel(
    const float* __restrict__ raw_sem, const float* __restrict__ Sigma,
    const float* __restrict__ mask, const float* __restrict__ Wgt,
    __bf16* __restrict__ F)
{
  __shared__ float red[4][64];
  __shared__ float geo[4][5];
  const int c  = threadIdx.x & 63;
  const int tl = threadIdx.x >> 6;
  const int m  = blockIdx.x * 4 + tl;

  if (c == 0) {
    const float* Sg = Sigma + (size_t)m * 9;
    const float s00 = Sg[0], s11 = Sg[4], s22 = Sg[8];
    const float s01 = 0.5f * (Sg[1] + Sg[3]);
    const float s02 = 0.5f * (Sg[2] + Sg[6]);
    const float s12 = 0.5f * (Sg[5] + Sg[7]);
    const float det = s00 * (s11 * s22 - s12 * s12)
                    - s01 * (s01 * s22 - s02 * s12)
                    + s02 * (s01 * s12 - s02 * s11);
    const float mk = mask[m];
    geo[tl][0] = __logf(fmaxf(det, EPSF)) * mk;
    geo[tl][1] = __logf(fmaxf(s00 + s11 + s22, EPSF)) * mk;
    geo[tl][2] = __logf(fmaxf(s00, EPSF)) * mk;
    geo[tl][3] = __logf(fmaxf(s11, EPSF)) * mk;
    geo[tl][4] = __logf(fmaxf(s22, EPSF)) * mk;
  }

  const float sv = raw_sem[(size_t)m * CGDIM + c];
  red[tl][c] = sv * sv;
  __syncthreads();
  for (int s = 32; s > 0; s >>= 1) {
    if (c < s) red[tl][c] += red[tl][c + s];
    __syncthreads();
  }
  const float nrm = fmaxf(sqrtf(red[tl][0]), NRMEPS);
  __syncthreads();
  F[(size_t)m * CF + c] = (__bf16)(sv / nrm);

  float g = 0.f;
#pragma unroll
  for (int j = 0; j < 5; ++j) g += geo[tl][j] * Wgt[c * 5 + j];
  red[tl][c] = g * g;
  __syncthreads();
  for (int s = 32; s > 0; s >>= 1) {
    if (c < s) red[tl][c] += red[tl][c + s];
    __syncthreads();
  }
  const float gn = fmaxf(sqrtf(red[tl][0]), NRMEPS);
  F[(size_t)m * CF + 64 + c] = (__bf16)(LAMG * g / gn);
}

// ---------------------------------------------------------------------------
// Bank features -> bf16: same structure for the 512 slots (no LAM, no mask).
// ---------------------------------------------------------------------------
__global__ __launch_bounds__(256) void bank_feat_kernel(
    const float* __restrict__ raw_sem, const float* __restrict__ gproto,
    const float* __restrict__ Wgb, __bf16* __restrict__ Bank)
{
  __shared__ float red[4][64];
  const int c  = threadIdx.x & 63;
  const int tl = threadIdx.x >> 6;
  const int k  = blockIdx.x * 4 + tl;

  const float sv = raw_sem[(size_t)k * CGDIM + c];
  red[tl][c] = sv * sv;
  __syncthreads();
  for (int s = 32; s > 0; s >>= 1) {
    if (c < s) red[tl][c] += red[tl][c + s];
    __syncthreads();
  }
  const float nrm = fmaxf(sqrtf(red[tl][0]), NRMEPS);
  __syncthreads();
  Bank[(size_t)k * CF + c] = (__bf16)(sv / nrm);

  float g = 0.f;
#pragma unroll
  for (int j = 0; j < 5; ++j) g += gproto[k * 5 + j] * Wgb[c * 5 + j];
  red[tl][c] = g * g;
  __syncthreads();
  for (int s = 32; s > 0; s >>= 1) {
    if (c < s) red[tl][c] += red[tl][c + s];
    __syncthreads();
  }
  const float gn = fmaxf(sqrtf(red[tl][0]), NRMEPS);
  Bank[(size_t)k * CF + 64 + c] = (__bf16)(g / gn);
}

// ---------------------------------------------------------------------------
// Greedy coverage: 1 persistent 512-thread block per batch (sim L2-resident).
// Thread t owns 4 consecutive k-columns (b128 loads) on an N-quarter:
//   kq = t & 127  -> columns 4*kq..4*kq+3,  part = t >> 7 -> n in [part*1024,..)
// Partial gain sums combined via LDS; masked tokens use covered=+BIG so the
// inner loop needs no mask test (gain contribution is exactly 0).
// ---------------------------------------------------------------------------
__global__ __launch_bounds__(512) void greedy_kernel(
    const float* __restrict__ sim, const float* __restrict__ mask,
    int* __restrict__ idx_out)
{
  if (__builtin_amdgcn_cluster_id_x() > 16) return;  // never true; cluster-id probe
  const int b = blockIdx.x;
  const float* S  = sim + (size_t)b * NTOK * KSLOT;
  const float* MK = mask + (size_t)b * NTOK;

  __shared__ float covered[NTOK];
  __shared__ float gp[4][KSLOT];
  __shared__ float gs[KSLOT];
  __shared__ int   gi[KSLOT];
  __shared__ int   picked[KSLOT];

  const int t    = threadIdx.x;
  const int kq   = t & 127;
  const int part = t >> 7;
  const int n0   = part * (NTOK / 4);
  const int n1   = n0 + (NTOK / 4);

  for (int n = t; n < NTOK; n += KSLOT)
    covered[n] = (MK[n] < 0.5f) ? BIGF : NEGF;
  picked[t] = 0;
  __syncthreads();

  const float4* colv = (const float4*)S + kq;   // stride KSLOT/4 float4 per row

  for (int it = 0; it < KUSE; ++it) {
    float g0 = 0.f, g1 = 0.f, g2 = 0.f, g3 = 0.f;
    __builtin_prefetch(colv + (size_t)n0 * (KSLOT / 4), 0, 1);
#pragma unroll 4
    for (int n = n0; n < n1; ++n) {
      const float4 sv = colv[(size_t)n * (KSLOT / 4)];
      const float cv = covered[n];
      g0 += fmaxf(cv, sv.x) - cv;
      g1 += fmaxf(cv, sv.y) - cv;
      g2 += fmaxf(cv, sv.z) - cv;
      g3 += fmaxf(cv, sv.w) - cv;
    }
    gp[part][kq * 4 + 0] = g0;
    gp[part][kq * 4 + 1] = g1;
    gp[part][kq * 4 + 2] = g2;
    gp[part][kq * 4 + 3] = g3;
    __syncthreads();

    gs[t] = picked[t] ? NEGF : (gp[0][t] + gp[1][t] + gp[2][t] + gp[3][t]);
    gi[t] = t;
    __syncthreads();

    // argmax (first-max on ties, matching jnp.argmax)
    for (int sd = KSLOT / 2; sd > 0; sd >>= 1) {
      if (t < sd) {
        const float v2 = gs[t + sd];
        const int   i2 = gi[t + sd];
        if (v2 > gs[t] || (v2 == gs[t] && i2 < gi[t])) { gs[t] = v2; gi[t] = i2; }
      }
      __syncthreads();
    }
    const int nxt = gi[0];
    if (t == 0) { picked[nxt] = 1; idx_out[b * KUSE + it] = nxt; }

    const float* scol = S + nxt;
    for (int n = t; n < NTOK; n += KSLOT)
      covered[n] = fmaxf(covered[n], scol[(size_t)n * KSLOT]);
    __syncthreads();
  }
}

// ---------------------------------------------------------------------------
// Gather + softmax: one block per token; the sim row (512 f32 = 2KB) is
// staged into LDS with the Tensor Data Mover, then gathered at LDS speed.
// ---------------------------------------------------------------------------
__global__ __launch_bounds__(512) void gather_softmax_kernel(
    const float* __restrict__ sim, const int* __restrict__ idxw,
    const float* __restrict__ mask, float* __restrict__ A0)
{
  __shared__ float row[KSLOT];
  __shared__ float red[KSLOT];
  const int bn = blockIdx.x;       // b*NTOK + n
  const int b  = bn >> 12;
  const int j  = threadIdx.x;
  const float* src = sim + (size_t)bn * KSLOT;

  if (threadIdx.x < 32) {          // wave 0 issues one TDM op for the block
    const uint64_t ga = (uint64_t)(uintptr_t)src;
    const uint32_t ldsoff = (uint32_t)(uintptr_t)(void*)row;
    u32x4 g0;
    g0.x = 1u;                                            // count=1, user D#
    g0.y = ldsoff;                                        // lds_addr (bytes)
    g0.z = (uint32_t)ga;                                  // global_addr lo
    g0.w = (uint32_t)((ga >> 32) & 0x1FFFFFFu) | (2u << 30); // addr hi | type=2
    i32x8 g1;
    g1[0] = (int)(2u << 16);       // data_size = 4B
    g1[1] = (int)(512u << 16);     // tensor_dim0 = 512 (lo16)
    g1[2] = (int)(1u << 16);       // tensor_dim1 = 1  (lo16)
    g1[3] = (int)(512u << 16);     // tile_dim0  = 512
    g1[4] = 1;                     // tile_dim1  = 1
    g1[5] = 512;                   // tensor_dim0_stride = 512 elements
    g1[6] = 0;
    g1[7] = 0;
    const i32x4 gz4 = {0, 0, 0, 0};
    const i32x8 gz8 = {0, 0, 0, 0, 0, 0, 0, 0};
    __builtin_amdgcn_tensor_load_to_lds(g0, g1, gz4, gz4, gz8, 0);
    __builtin_amdgcn_s_wait_tensorcnt((short)0);
  }
  __syncthreads();

  const float mkv = mask[bn];
  const int   k   = idxw[b * KUSE + j];
  const float v   = row[k & (KSLOT - 1)];
  const float logit = (mkv < 0.5f) ? NEGF : v * TAUINV;

  red[j] = logit;
  __syncthreads();
  for (int s = KSLOT / 2; s > 0; s >>= 1) {
    if (j < s) red[j] = fmaxf(red[j], red[j + s]);
    __syncthreads();
  }
  const float mx = red[0];
  __syncthreads();
  const float e = __expf(logit - mx);
  red[j] = e;
  __syncthreads();
  for (int s = KSLOT / 2; s > 0; s >>= 1) {
    if (j < s) red[j] += red[j + s];
    __syncthreads();
  }
  const float sum = red[0];
  A0[(size_t)bn * KSLOT + j] = (e / sum) * mkv;
}

// ---------------------------------------------------------------------------
// Moment accumulation for soft Gaussian merge.  One thread per (b,k):
// S0, S1=sum A*mu, S2=sum A*(sym(Sigma)+mu mu^T); then
// Sig_k = S2/occ - mu_k mu_k^T + JIT*I  (algebraically == reference).
// Also emits occ and idx (as float) into d_out.
// ---------------------------------------------------------------------------
__global__ __launch_bounds__(256) void moments_kernel(
    const float* __restrict__ A0, const float* __restrict__ mu,
    const float* __restrict__ Sigma, const int* __restrict__ idxw,
    float* __restrict__ out)
{
  const int t = blockIdx.x * blockDim.x + threadIdx.x;
  if (t >= BB * KSLOT) return;
  const int b = t >> 9;
  const int k = t & (KSLOT - 1);
  const float* Ab  = A0 + (size_t)b * NTOK * KSLOT + k;
  const float* mub = mu + (size_t)b * NTOK * 3;
  const float* Sb  = Sigma + (size_t)b * NTOK * 9;

  float s0 = 0.f, s1x = 0.f, s1y = 0.f, s1z = 0.f;
  float sxx = 0.f, sxy = 0.f, sxz = 0.f, syy = 0.f, syz = 0.f, szz = 0.f;
  for (int n = 0; n < NTOK; ++n) {
    const float a  = Ab[(size_t)n * KSLOT];
    const float mx = mub[n * 3 + 0], my = mub[n * 3 + 1], mz = mub[n * 3 + 2];
    const float* Sg = Sb + (size_t)n * 9;
    s0  += a;
    s1x += a * mx; s1y += a * my; s1z += a * mz;
    sxx += a * (Sg[0] + mx * mx);
    syy += a * (Sg[4] + my * my);
    szz += a * (Sg[8] + mz * mz);
    sxy += a * (0.5f * (Sg[1] + Sg[3]) + mx * my);
    sxz += a * (0.5f * (Sg[2] + Sg[6]) + mx * mz);
    syz += a * (0.5f * (Sg[5] + Sg[7]) + my * mz);
  }
  const float occ = fmaxf(s0, EPSF);
  const float inv = 1.f / occ;
  const float mkx = s1x * inv, mky = s1y * inv, mkz = s1z * inv;
  const float g00 = sxx * inv - mkx * mkx + JIT;
  const float g11 = syy * inv - mky * mky + JIT;
  const float g22 = szz * inv - mkz * mkz + JIT;
  const float g01 = sxy * inv - mkx * mky;
  const float g02 = sxz * inv - mkx * mkz;
  const float g12 = syz * inv - mky * mkz;

  float* muo  = out + 4194304;            // after A0 [2*4096*512]
  float* sigo = out + 4197376;            // + mu_k  [2*512*3]
  float* occo = out + 4206592;            // + Sig_k [2*512*9]
  float* idxo = out + 4207616;            // + occ   [2*512]
  muo[t * 3 + 0] = mkx; muo[t * 3 + 1] = mky; muo[t * 3 + 2] = mkz;
  float* S9 = sigo + (size_t)t * 9;
  S9[0] = g00; S9[1] = g01; S9[2] = g02;
  S9[3] = g01; S9[4] = g11; S9[5] = g12;
  S9[6] = g02; S9[7] = g12; S9[8] = g22;
  occo[t] = occ;
  idxo[t] = (float)idxw[t];
}

// ---------------------------------------------------------------------------
extern "C" void kernel_launch(void* const* d_in, const int* in_sizes, int n_in,
                              void* d_out, int out_size, void* d_ws, size_t ws_size,
                              hipStream_t stream)
{
  const float* s    = (const float*)d_in[0];  // [2,4096,512]
  const float* mu   = (const float*)d_in[1];  // [2,4096,3]
  const float* Sig  = (const float*)d_in[2];  // [2,4096,3,3]
  const float* mask = (const float*)d_in[3];  // [2,4096]
  const float* slot = (const float*)d_in[4];  // [512,512]
  const float* gpro = (const float*)d_in[5];  // [512,5]
  const float* Wgt  = (const float*)d_in[6];  // [64,5]
  const float* Wgb  = (const float*)d_in[7];  // [64,5]
  const float* Wst  = (const float*)d_in[8];  // [64,512]
  const float* Wsb  = (const float*)d_in[9];  // [64,512]
  float* out = (float*)d_out;
  char*  w   = (char*)d_ws;

  // workspace layout (byte offsets, all 16B aligned)
  float*  sim     = (float*)(w);                 // 2*4096*512 f32 = 16 MB
  float*  rawT    = (float*)(w + 16777216);      // 8192*64  f32   =  2 MB
  float*  rawB    = (float*)(w + 18874368);      // 512*64   f32   = 128 KB
  __bf16* s_bf    = (__bf16*)(w + 19005440);     // 8192*512 bf16  =  8 MB
  __bf16* slot_bf = (__bf16*)(w + 27394048);     // 512*512  bf16  = 512 KB
  __bf16* F_bf    = (__bf16*)(w + 27918336);     // 8192*128 bf16  =  2 MB
  __bf16* Bank_bf = (__bf16*)(w + 30015488);     // 512*128  bf16  = 128 KB
  int*    idxw    = (int*)(w + 30146560);        // 2*512 int
  __bf16* Wst_bf  = (__bf16*)(w + 30150656);     // 64*512 bf16
  __bf16* Wsb_bf  = (__bf16*)(w + 30216192);     // 64*512 bf16

  // 0) one-time bf16 materialization of GEMM operands
  cvt_bf16_kernel<<<(MTOK * CDIM + 255) / 256, 256, 0, stream>>>(s, s_bf, MTOK * CDIM);
  cvt_bf16_kernel<<<(KSLOT * CDIM + 255) / 256, 256, 0, stream>>>(slot, slot_bf, KSLOT * CDIM);
  cvt_bf16_kernel<<<(CGDIM * CDIM + 255) / 256, 256, 0, stream>>>(Wst, Wst_bf, CGDIM * CDIM);
  cvt_bf16_kernel<<<(CGDIM * CDIM + 255) / 256, 256, 0, stream>>>(Wsb, Wsb_bf, CGDIM * CDIM);

  // 1) raw semantic projections (WMMA bf16 GEMMs, f32 accumulate)
  {
    const int waves = (MTOK / 16) * (CGDIM / 64);              // 512
    wmma_gemm_bf16<<<(waves * 32 + 255) / 256, 256, 0, stream>>>(
        s_bf, Wst_bf, rawT, MTOK, CGDIM, CDIM);
  }
  {
    const int waves = (KSLOT / 16) * (CGDIM / 64);             // 32
    wmma_gemm_bf16<<<(waves * 32 + 255) / 256, 256, 0, stream>>>(
        slot_bf, Wsb_bf, rawB, KSLOT, CGDIM, CDIM);
  }

  // 2) normalize + geo path -> combined 128-wide bf16 features
  token_feat_kernel<<<MTOK / 4, 256, 0, stream>>>(rawT, Sig, mask, Wgt, F_bf);
  bank_feat_kernel<<<KSLOT / 4, 256, 0, stream>>>(rawB, gpro, Wgb, Bank_bf);

  // 3) sim = F @ Bank^T  (folds sem + LAM*geo in one WMMA GEMM)
  {
    const int waves = (MTOK / 16) * (KSLOT / 64);              // 4096
    wmma_gemm_bf16<<<(waves * 32 + 255) / 256, 256, 0, stream>>>(
        F_bf, Bank_bf, sim, MTOK, KSLOT, CF);
  }

  // 4) sequential greedy coverage (one persistent block per batch)
  greedy_kernel<<<BB, 512, 0, stream>>>(sim, mask, idxw);

  // 5) gather + softmax -> A0 (TDM-staged rows)
  gather_softmax_kernel<<<BB * NTOK, 512, 0, stream>>>(sim, idxw, mask, out);

  // 6) Gaussian merge moments + tail outputs
  moments_kernel<<<(BB * KSLOT + 255) / 256, 256, 0, stream>>>(
      out, mu, Sig, idxw, out);

  (void)in_sizes; (void)n_in; (void)out_size; (void)ws_size;
}
